// MixHopGCN_15221364097510
// MI455X (gfx1250) — matrix operand order repaired and checked by reading, compiled, not run on hardware
//
#include <hip/hip_runtime.h>

typedef __attribute__((ext_vector_type(16))) _Float16 v16h;
typedef __attribute__((ext_vector_type(8)))  _Float16 v8h;
typedef __attribute__((ext_vector_type(8)))  float    v8f;

#define NN    100000
#define EE    1600000
#define DIN   64
#define HID   512
#define CATD  192      // 3 * 64 fused hidden inputs
#define OUTD  64
#define BN_EPS 1e-5f

#define NT        (NN / 16)   // 6250 row tiles
#define GEMM_GX   256         // row-blocks (grid-stride over tiles)
#define BCOLS     256         // columns staged per block (half of HID)
#define LDS_STR   200         // padded col stride in halves (bank-conflict-free)

// ---------------- degree / norm ----------------
__global__ __launch_bounds__(256) void k_deg(const int* __restrict__ dst,
                                             float* __restrict__ deg) {
  int e = blockIdx.x * 256 + threadIdx.x;
  if (e < EE) atomicAdd(&deg[dst[e]], 1.0f);
}

__global__ __launch_bounds__(256) void k_norm(const float* __restrict__ deg,
                                              float* __restrict__ nrm) {
  int i = blockIdx.x * 256 + threadIdx.x;
  if (i < NN) {
    float d = deg[i];
    d = d < 1.0f ? 1.0f : d;
    nrm[i] = rsqrtf(d);
  }
}

// ---------------- edge scatter: acc[dst] += h[src] * nrm[src] ----------------
// 16 threads per edge, 4 features each. Feature matrix is 25.6MB -> L2-resident,
// so gathers and float atomics stay on-chip.
__global__ __launch_bounds__(256) void k_scatter(const float* __restrict__ h,
                                                 const float* __restrict__ nrm,
                                                 const int* __restrict__ src,
                                                 const int* __restrict__ dst,
                                                 float* __restrict__ acc) {
  int id = blockIdx.x * 256 + threadIdx.x;   // EE*16 threads exactly
  int e  = id >> 4;
  int f  = (id & 15) << 2;
  int s = src[e];
  int d = dst[e];
  float w = nrm[s];
  const float4 v = *(const float4*)(h + (size_t)s * DIN + f);
  float* a = acc + (size_t)d * DIN + f;
  atomicAdd(a + 0, v.x * w);
  atomicAdd(a + 1, v.y * w);
  atomicAdd(a + 2, v.z * w);
  atomicAdd(a + 3, v.w * w);
}

// post-scale by nrm[dst]; keep fp32 for next hop, store f16 into Hcat column block j
__global__ __launch_bounds__(256) void k_scale(const float* __restrict__ acc,
                                               const float* __restrict__ nrm,
                                               float* __restrict__ hnext,
                                               _Float16* __restrict__ hcat, int j) {
  int id = blockIdx.x * 256 + threadIdx.x;   // NN*64 threads exactly
  int i = id >> 6, f = id & 63;
  float v = acc[id] * nrm[i];
  hnext[id] = v;
  hcat[(size_t)i * CATD + j * 64 + f] = (_Float16)v;
}

// ---------------- weight fusion: M[f, j*64+c] = sum_s fc_W[f, j*512+s] * Wj[s,c] ----------------
__global__ __launch_bounds__(256) void k_precompM(const float* __restrict__ fcW,
                                                  const float* __restrict__ W0,
                                                  const float* __restrict__ W1,
                                                  const float* __restrict__ W2,
                                                  _Float16* __restrict__ M) {
  int id = blockIdx.x * 256 + threadIdx.x;   // 512*192 threads
  int f = id / CATD, c = id % CATD;
  int j = c >> 6, cc = c & 63;
  const float* Wj = (j == 0) ? W0 : (j == 1 ? W1 : W2);
  const float* fr = fcW + (size_t)f * (3 * HID) + j * HID;
  float s = 0.0f;
  for (int k = 0; k < HID; ++k) s += fr[k] * Wj[k * 64 + cc];
  M[(size_t)f * CATD + c] = (_Float16)s;   // row-major (512 x 192): K contiguous per col
}

// bias_eff[f] = fc_b[f] + sum_j Fj @ bj
__global__ __launch_bounds__(256) void k_biaseff(const float* __restrict__ fcW,
                                                 const float* __restrict__ fcb,
                                                 const float* __restrict__ b0,
                                                 const float* __restrict__ b1,
                                                 const float* __restrict__ b2,
                                                 float* __restrict__ beff) {
  int f = blockIdx.x * 256 + threadIdx.x;
  if (f >= HID) return;
  const float* fr = fcW + (size_t)f * (3 * HID);
  float s = fcb[f];
  for (int k = 0; k < HID; ++k)
    s += fr[k] * b0[k] + fr[HID + k] * b1[k] + fr[2 * HID + k] * b2[k];
  beff[f] = s;
}

// ---------------- fused GEMM: final(N x 512) = Hcat(N x 192) @ M^T + bias_eff ----------------
// Block = 8 waves; block stages a 256-column half of M (f16) into LDS once
// (padded stride 200 halves -> conflict-free ds_load of B fragments), then
// grid-strides over ~25 row tiles reusing LDS-resident B. Wave owns 32 cols
// (2 accumulators); A fragments come from global (16 rows x 384B, WGP$-hot).
//
// Fragment layouts used here (contiguous k order, same for A and B):
//  A (f16 16x32): lane row = lane%16; half i <-> K = kb+16*(lane/16)+i
//                 (two 16B loads)
//  B (f16 32x16): lane col = lane%16; half i <-> K = kb+16*(lane/16)+i
//                 (one contiguous 32B LDS read at col*LDS_STR + kb + 16*hf)
//  C/D (f32 16x16): VGPR r <-> row r + 8*(lane/16), col = lane%16
__global__ __launch_bounds__(256) void k_gemm(const _Float16* __restrict__ A,
                                              const _Float16* __restrict__ B,
                                              const float* __restrict__ bias,
                                              float* __restrict__ out) {
  __shared__ _Float16 Bs[BCOLS * LDS_STR];   // 102,400 bytes (<< 320KB WGP LDS)

  const int tid  = threadIdx.x;
  const int wave = tid >> 5;
  const int lane = tid & 31;
  const int hf   = lane >> 4;
  const int lm   = lane & 15;
  const int colBase = blockIdx.y * BCOLS;

  // ---- stage this block's half of M into LDS (256 cols x first 96 K) ----
#pragma unroll
  for (int it = 0; it < 12; ++it) {          // 3072 16-byte chunks / 256 threads
    int idx  = it * 256 + tid;
    int col  = idx / 12;
    int part = idx % 12;
    v8h v = *(const v8h*)(B + (size_t)(colBase + col) * CATD + part * 8);
    *(v8h*)(&Bs[col * LDS_STR + part * 8]) = v;
  }
  __syncthreads();

  // wave covers cols [colBase + wave*32, +32): two 16-col tiles
  const int cl0 = wave * 32 + lm;            // local col of tile 0
  const _Float16* Bl0 = &Bs[cl0 * LDS_STR + hf * 16];
  const _Float16* Bl1 = Bl0 + 16 * LDS_STR;
  const int col0 = colBase + cl0;
  const float bi0 = bias[col0];
  const float bi1 = bias[col0 + 16];

  for (int rt = blockIdx.x; rt < NT; rt += GEMM_GX) {
    const _Float16* Arow = A + (size_t)(rt * 16 + lm) * CATD;
    v8f c0 = {}, c1 = {};
#pragma unroll
    for (int kb = 0; kb < CATD; kb += 32) {
      v8h alo = *(const v8h*)(Arow + kb + 16 * hf);
      v8h ahi = *(const v8h*)(Arow + kb + 16 * hf + 8);
      v16h a = __builtin_shufflevector(alo, ahi, 0, 1, 2, 3, 4, 5, 6, 7,
                                                 8, 9, 10, 11, 12, 13, 14, 15);
      v16h b0v = *(const v16h*)(Bl0 + kb);
      v16h b1v = *(const v16h*)(Bl1 + kb);
      c0 = __builtin_amdgcn_wmma_f32_16x16x32_f16(false, a, false, b0v, (short)0, c0, false, false);
      c1 = __builtin_amdgcn_wmma_f32_16x16x32_f16(false, a, false, b1v, (short)0, c1, false, false);
    }
#pragma unroll
    for (int r = 0; r < 8; ++r) {
      size_t ro = (size_t)(rt * 16 + r + 8 * hf) * HID;
      out[ro + col0]      = c0[r] + bi0;
      out[ro + col0 + 16] = c1[r] + bi1;
    }
  }
}

// ---------------- BN stats: per-column sum & sumsq over N rows ----------------
__global__ __launch_bounds__(256) void k_bnstats(const float* __restrict__ fin,
                                                 float* __restrict__ sums,
                                                 float* __restrict__ sumsq) {
  __shared__ float sh1[256];
  __shared__ float sh2[256];
  int tx = threadIdx.x & 63;        // column offset
  int ty = threadIdx.x >> 6;        // row sub-group 0..3
  int col = blockIdx.x * 64 + tx;
  float s = 0.0f, s2 = 0.0f;
  for (int r = blockIdx.y * 4 + ty; r < NN; r += gridDim.y * 4) {
    float v = fin[(size_t)r * HID + col];
    s += v;
    s2 += v * v;
  }
  sh1[threadIdx.x] = s;
  sh2[threadIdx.x] = s2;
  __syncthreads();
  if (ty == 0) {
    s  = sh1[tx] + sh1[tx + 64] + sh1[tx + 128] + sh1[tx + 192];
    s2 = sh2[tx] + sh2[tx + 64] + sh2[tx + 128] + sh2[tx + 192];
    atomicAdd(&sums[col], s);
    atomicAdd(&sumsq[col], s2);
  }
}

// coefA[c] = rs*gamma*q[k];  coefB[c] = (beta - mu*rs*gamma)*q[k]
__global__ __launch_bounds__(256) void k_coef(const float* __restrict__ sums,
                                              const float* __restrict__ sumsq,
                                              const float* __restrict__ gamma,
                                              const float* __restrict__ beta,
                                              const float* __restrict__ q,
                                              float* __restrict__ coefA,
                                              float* __restrict__ coefB) {
  int c = blockIdx.x * 256 + threadIdx.x;
  if (c >= HID) return;
  const float invN = 1.0f / (float)NN;
  float mu  = sums[c] * invN;
  float var = sumsq[c] * invN - mu * mu;
  float g   = rsqrtf(var + BN_EPS) * gamma[c];
  float qq  = q[c >> 6];
  coefA[c] = g * qq;
  coefB[c] = (beta[c] - mu * g) * qq;
}

__global__ void k_consto(const float* __restrict__ coefB, float* __restrict__ constO) {
  int o = threadIdx.x;               // 64 threads
  float s = 0.0f;
#pragma unroll
  for (int k = 0; k < 8; ++k) s += coefB[k * 64 + o];
  constO[o] = s;
}

// out[n,o] = sum_k final[n, k*64+o]*coefA[k*64+o] + constO[o]
__global__ __launch_bounds__(256) void k_outq(const float* __restrict__ fin,
                                              const float* __restrict__ coefA,
                                              const float* __restrict__ constO,
                                              float* __restrict__ out) {
  int id = blockIdx.x * 256 + threadIdx.x;   // NN*64 threads exactly
  int n = id >> 6, o = id & 63;
  const float* f = fin + (size_t)n * HID + o;
  float s = constO[o];
#pragma unroll
  for (int k = 0; k < 8; ++k) s += f[k * 64] * coefA[k * 64 + o];
  out[id] = s;
}

extern "C" void kernel_launch(void* const* d_in, const int* in_sizes, int n_in,
                              void* d_out, int out_size, void* d_ws, size_t ws_size,
                              hipStream_t stream) {
  const float* feats = (const float*)d_in[0];
  const int*   src   = (const int*)d_in[1];
  const int*   dst   = (const int*)d_in[2];
  const float* W0    = (const float*)d_in[3];
  const float* b0    = (const float*)d_in[4];
  const float* W1    = (const float*)d_in[5];
  const float* b1    = (const float*)d_in[6];
  const float* W2    = (const float*)d_in[7];
  const float* b2    = (const float*)d_in[8];
  const float* fcW   = (const float*)d_in[9];
  const float* fcb   = (const float*)d_in[10];
  const float* gamma = (const float*)d_in[11];
  const float* beta  = (const float*)d_in[12];
  const float* q     = (const float*)d_in[13];

  // workspace carve-out (256B aligned)
  size_t off = 0;
  auto alloc = [&](size_t bytes) -> void* {
    off = (off + 255) & ~(size_t)255;
    void* p = (char*)d_ws + off;
    off += bytes;
    return p;
  };
  float*     deg    = (float*)alloc((size_t)NN * 4);
  float*     nrm    = (float*)alloc((size_t)NN * 4);
  float*     hcur   = (float*)alloc((size_t)NN * DIN * 4);
  float*     hacc   = (float*)alloc((size_t)NN * DIN * 4);
  _Float16*  hcat   = (_Float16*)alloc((size_t)NN * CATD * 2);
  _Float16*  Mmat   = (_Float16*)alloc((size_t)HID * CATD * 2);
  float*     beff   = (float*)alloc((size_t)HID * 4);
  float*     fin    = (float*)alloc((size_t)NN * HID * 4);
  float*     sums   = (float*)alloc((size_t)HID * 4);
  float*     sumsq  = (float*)alloc((size_t)HID * 4);
  float*     coefA  = (float*)alloc((size_t)HID * 4);
  float*     coefB  = (float*)alloc((size_t)HID * 4);
  float*     constO = (float*)alloc((size_t)OUTD * 4);
  (void)ws_size; (void)in_sizes; (void)n_in; (void)out_size;

  // degree + normalization
  hipMemsetAsync(deg, 0, (size_t)NN * 4, stream);
  k_deg<<<(EE + 255) / 256, 256, 0, stream>>>(dst, deg);
  k_norm<<<(NN + 255) / 256, 256, 0, stream>>>(deg, nrm);

  // weight fusion (independent of graph work; overlaps naturally)
  k_precompM<<<(HID * CATD) / 256, 256, 0, stream>>>(fcW, W0, W1, W2, Mmat);
  k_biaseff<<<2, 256, 0, stream>>>(fcW, fcb, b0, b1, b2, beff);

  // 3 propagation hops; hop j fills Hcat column block j
  for (int j = 0; j < 3; ++j) {
    hipMemsetAsync(hacc, 0, (size_t)NN * DIN * 4, stream);
    const float* hin = (j == 0) ? feats : hcur;
    k_scatter<<<(EE * 16) / 256, 256, 0, stream>>>(hin, nrm, src, dst, hacc);
    k_scale<<<(NN * 64) / 256, 256, 0, stream>>>(hacc, nrm, hcur, hcat, j);
  }

  // one fused WMMA GEMM for all three linears + fc layer (LDS-staged weights)
  k_gemm<<<dim3(GEMM_GX, HID / BCOLS), 256, 0, stream>>>(hcat, Mmat, beff, fin);

  // batch-norm stats + fused BN*q output map
  hipMemsetAsync(sums, 0, (size_t)HID * 4, stream);
  hipMemsetAsync(sumsq, 0, (size_t)HID * 4, stream);
  k_bnstats<<<dim3(HID / 64, 256), 256, 0, stream>>>(fin, sums, sumsq);
  k_coef<<<2, 256, 0, stream>>>(sums, sumsq, gamma, beta, q, coefA, coefB);
  k_consto<<<1, 64, 0, stream>>>(coefB, constO);
  k_outq<<<(NN * 64) / 256, 256, 0, stream>>>(fin, coefA, constO, (float*)d_out);
}
